// ViT_Custom_Model_72249939853848
// MI455X (gfx1250) — compile-verified, matrix operand order checked by reading, not run on hardware
//
#include <hip/hip_runtime.h>
#include <math.h>

// ---------------------------------------------------------------------------
// ViT forward for MI455X (gfx1250).
// - All GEMMs: v_wmma_f32_16x16x32_f16, fp32 accumulate, f16 operands.
// - Weights pre-transposed+converted to f16 B^T layout once per call.
// - Activations kept f16 in workspace; biases/LN stats fp32.
// - Tile staging via double-buffered TDM (tensor_load_to_lds, TENSORcnt
//   pipelined with s_wait_tensorcnt 2) when the builtin exists; manual
//   vectorized staging otherwise.
// Block = 256 threads = 8 wave32 waves (4x2), block tile 128x128,
// wave tile 32x64 (8 WMMA per K-step), K-step 32.
// ---------------------------------------------------------------------------

typedef _Float16 half_t;
typedef __attribute__((ext_vector_type(16))) _Float16 v16h;
typedef __attribute__((ext_vector_type(8)))  _Float16 v8h;
typedef __attribute__((ext_vector_type(8)))  float    v8f;
typedef __attribute__((ext_vector_type(4)))  unsigned int u32x4;
typedef __attribute__((ext_vector_type(8)))  int i32x8;
typedef __attribute__((ext_vector_type(4)))  int i32x4;

#define TILE_R 128   // block rows
#define TILE_C 128   // block cols
#define TILE_K 32
#define LDSS   40    // padded LDS row stride (halves): 32 data + 8 pad
#define NPAD   208   // padded pitch for 196-wide f16 activations (16B aligned)

#ifndef __has_builtin
#define __has_builtin(x) 0
#endif
#if __has_builtin(__builtin_amdgcn_tensor_load_to_lds) && \
    __has_builtin(__builtin_amdgcn_s_wait_tensorcnt)
#define HAVE_TDM 1
#else
#define HAVE_TDM 0
#endif

enum {
  EPI_PLAIN = 0,
  EPI_PATCH,      // + bias1[col] + bias2[(row%modrow)*ldb2+col]        -> f16
  EPI_QKV,        // + bias1[col], store [B,H,N,M]                      -> f16
  EPI_QKVT,       // + bias1[col], store [B,H,M,NPAD] (transposed V)    -> f16
  EPI_SCORES,     // * scale, batched [N,NPAD]                          -> f16
  EPI_ATTNO,      // store o[b, m, h*N+n]                               -> f16
  EPI_OW1T,       // + bias2[row*ldb2+col], store [b][col][row]         -> f16
  EPI_OW2,        // + bias2[(row%modrow)*ldb2+col] + resid             -> f16
  EPI_GELU,       // gelu(acc + bias1[col])                             -> f16
  EPI_GELU_RES,   // gelu(acc + bias1[col]) + resid                     -> f16
  EPI_BIAS1       // acc + bias1[col]                                   -> f32
};

__device__ __forceinline__ float gelu_exact(float x) {
  return 0.5f * x * (1.0f + erff(x * 0.70710678118654752f));
}

__device__ __forceinline__ v16h ld_frag(const half_t* rowp, int koff) {
  v8h lo = *(const v8h*)(rowp + koff);
  v8h hi = *(const v8h*)(rowp + koff + 16);
  return __builtin_shufflevector(lo, hi, 0,1,2,3,4,5,6,7,8,9,10,11,12,13,14,15);
}

#if HAVE_TDM
// Issue one TDM 2D tile load: TILE_R rows x 32 halves from row-major f16
// source (row stride = strideElems) into LDS rows padded to LDSS halves.
// tensor_dim clamps (colsLeft/rowsLeft) give zero fill at K / row edges.
__device__ __forceinline__ void tdm_load_tile(half_t* ldsDst, const half_t* gsrc,
                                              int strideElems, int rowsLeft,
                                              int colsLeft) {
  unsigned lds_addr = (unsigned)(size_t)ldsDst;       // low 32 bits = LDS offset
  unsigned long long ga = (unsigned long long)(size_t)gsrc;
  int td0 = colsLeft < 0 ? 0 : colsLeft;              // valid K extent
  int td1 = rowsLeft < 0 ? 0 : rowsLeft;              // valid row extent
  u32x4 g0;
  g0[0] = 1u;                                         // count=1, no gather
  g0[1] = lds_addr;
  g0[2] = (unsigned)ga;                               // global_addr[31:0]
  g0[3] = ((unsigned)(ga >> 32) & 0x01FFFFFFu) | (2u << 30);  // [56:32] | type=2
  i32x8 g1;
  // data_size=2B(enc 1), pad_enable, pad_interval=16 DWORDs(enc 3),
  // pad_amount=4 DWORDs(enc 3), workgroup_mask=0
  g1[0] = (int)((1u << 16) | (1u << 20) | (3u << 22) | (3u << 25));
  g1[1] = (int)(((unsigned)td0 & 0xFFFFu) << 16);                 // tensor_dim0 lo
  g1[2] = (int)((((unsigned)td0 >> 16) & 0xFFFFu) |
                (((unsigned)td1 & 0xFFFFu) << 16));               // dim0 hi | dim1 lo
  g1[3] = (int)((((unsigned)td1 >> 16) & 0xFFFFu) |
                ((unsigned)TILE_K << 16));                        // dim1 hi | tile_dim0
  g1[4] = TILE_R;                                                 // tile_dim1 (tile_dim2=0)
  g1[5] = strideElems;                                            // tensor_dim0_stride lo
  g1[6] = 0;
  g1[7] = 0;
  i32x4 gz = {0, 0, 0, 0};
#if defined(__clang_major__) && (__clang_major__ >= 23)
  i32x8 gz8 = {0, 0, 0, 0, 0, 0, 0, 0};
  __builtin_amdgcn_tensor_load_to_lds(g0, g1, gz, gz, gz8, 0);
#else
  __builtin_amdgcn_tensor_load_to_lds(g0, g1, gz, gz, 0);
#endif
}
#endif  // HAVE_TDM

// Manual staging (fallback): both A and B are row-major [rows x K] f16.
__device__ __forceinline__ void stage_tile(half_t dst[TILE_R][LDSS],
                                           const half_t* __restrict__ src,
                                           int rows, int rowBase, int ld,
                                           int kk, int K, int tid) {
  bool full = (rowBase + TILE_R <= rows) && (kk + TILE_K <= K);
  if (full) {
    #pragma unroll
    for (int it = 0; it < 2; ++it) {
      int i  = tid + it * 256;        // 512 chunks: 128 rows x 4 chunks
      int r  = i >> 2;
      int c8 = (i & 3) << 3;
      *(v8h*)&dst[r][c8] =
          *(const v8h*)(src + (long long)(rowBase + r) * ld + kk + c8);
    }
  } else {
    #pragma unroll
    for (int it = 0; it < 2; ++it) {
      int i  = tid + it * 256;
      int r  = i >> 2;
      int c8 = (i & 3) << 3;
      int gr = rowBase + r;
      #pragma unroll
      for (int u = 0; u < 8; ++u) {
        half_t v = (half_t)0.f;
        if (gr < rows && kk + c8 + u < K)
          v = src[(long long)gr * ld + kk + c8 + u];
        dst[r][c8 + u] = v;
      }
    }
  }
}

template <int EPI>
__global__ __launch_bounds__(256)
void gemm_wmma(const half_t* __restrict__ A, const half_t* __restrict__ Bm,
               void* __restrict__ CoutV,
               const float* __restrict__ bias1, const float* __restrict__ bias2,
               const half_t* __restrict__ resid,
               int Ra, int K, int Ncols,
               int lda, long long sA, int ldb, long long sB,
               int ldc, long long sC,
               int modrow, int ldb2, float scale,
               int Hh, int Mm, int Nn) {
  __shared__ half_t As[2][TILE_R][LDSS];
  __shared__ half_t Bs[2][TILE_R][LDSS];

  const int tid  = threadIdx.x;
  const int wave = tid >> 5;
  const int lane = tid & 31;
  const int wr   = wave >> 1;   // 0..3 (row group, 32 rows each)
  const int wc   = wave & 1;    // 0..1 (col group, 64 cols each)
  const int z    = blockIdx.z;
  const int rowBase = blockIdx.y * TILE_R;
  const int colBase = blockIdx.x * TILE_C;

  const half_t* Ab = A  + (long long)z * sA;
  const half_t* Bb = Bm + (long long)z * sB;

  v8f acc[2][4] = {};
  const int fr   = lane & 15;
  const int koff = (lane >> 4) << 3;   // 0 or 8

  auto compute_step = [&](const half_t (&At)[TILE_R][LDSS],
                          const half_t (&Bt)[TILE_R][LDSS]) {
    v16h af[2], bf[4];
    #pragma unroll
    for (int t = 0; t < 2; ++t)
      af[t] = ld_frag(&At[wr * 32 + t * 16 + fr][0], koff);
    #pragma unroll
    for (int t = 0; t < 4; ++t)
      bf[t] = ld_frag(&Bt[wc * 64 + t * 16 + fr][0], koff);
    #pragma unroll
    for (int i = 0; i < 2; ++i)
      #pragma unroll
      for (int j = 0; j < 4; ++j)
        acc[i][j] = __builtin_amdgcn_wmma_f32_16x16x32_f16(
            false, af[i], false, bf[j], (short)0, acc[i][j], false, false);
  };

#if HAVE_TDM
  // --- double-buffered TDM pipeline: overlap tile k+1 DMA with tile k WMMAs
  const int nk = (K + TILE_K - 1) / TILE_K;
  if (wave == 0) {
    tdm_load_tile(&As[0][0][0], Ab + (long long)rowBase * lda, lda,
                  Ra - rowBase, K);
    tdm_load_tile(&Bs[0][0][0], Bb + (long long)colBase * ldb, ldb,
                  Ncols - colBase, K);
  }
  for (int kt = 0; kt < nk; ++kt) {
    const int cur = kt & 1;
    const int kk  = kt * TILE_K;
    if (wave == 0) {
      if (kk + TILE_K < K) {
        tdm_load_tile(&As[cur ^ 1][0][0],
                      Ab + (long long)rowBase * lda + kk + TILE_K, lda,
                      Ra - rowBase, K - kk - TILE_K);
        tdm_load_tile(&Bs[cur ^ 1][0][0],
                      Bb + (long long)colBase * ldb + kk + TILE_K, ldb,
                      Ncols - colBase, K - kk - TILE_K);
        // TENSORcnt decrements in order: <=2 outstanding means tile kt landed.
        __builtin_amdgcn_s_wait_tensorcnt((short)2);
      } else {
        __builtin_amdgcn_s_wait_tensorcnt((short)0);
      }
    }
    __syncthreads();                 // buf[cur] visible to all waves
    compute_step(As[cur], Bs[cur]);
    __syncthreads();                 // reads done; buf[cur] may be overwritten
  }
#else
  for (int kk = 0; kk < K; kk += TILE_K) {
    stage_tile(As[0], Ab, Ra, rowBase, lda, kk, K, tid);
    stage_tile(Bs[0], Bb, Ncols, colBase, ldb, kk, K, tid);
    if (kk + TILE_K < K) {
      int pr = rowBase + (tid >> 1);
      if (pr < Ra) __builtin_prefetch(Ab + (long long)pr * lda + kk + TILE_K, 0, 3);
    }
    __syncthreads();
    compute_step(As[0], Bs[0]);
    __syncthreads();
  }
#endif

  // ---- epilogue (C fragment: VGPR e, lane -> row e+8*(lane>=16), col lane&15)
  const int rsel = (lane >> 4) << 3;
  half_t* cb = (half_t*)CoutV + (long long)z * sC;
  #pragma unroll
  for (int ti = 0; ti < 2; ++ti) {
    #pragma unroll
    for (int tj = 0; tj < 4; ++tj) {
      #pragma unroll
      for (int e = 0; e < 8; ++e) {
        int r = rowBase + wr * 32 + ti * 16 + rsel + e;
        int c = colBase + wc * 64 + tj * 16 + fr;
        if (r < Ra && c < Ncols) {
          float v = acc[ti][tj][e];
          if constexpr (EPI == EPI_PLAIN) {
            cb[(long long)r * ldc + c] = (half_t)v;
          } else if constexpr (EPI == EPI_PATCH) {
            v += bias1[c] + bias2[(r % modrow) * ldb2 + c];
            cb[(long long)r * ldc + c] = (half_t)v;
          } else if constexpr (EPI == EPI_QKV) {
            v += bias1[c];
            int b = r / Nn, n = r % Nn, m = c / Hh, h = c % Hh;
            ((half_t*)CoutV)[((((long long)b * Hh + h) * Nn + n) * Mm) + m] = (half_t)v;
          } else if constexpr (EPI == EPI_QKVT) {
            v += bias1[c];
            int b = r / Nn, n = r % Nn, m = c / Hh, h = c % Hh;
            ((half_t*)CoutV)[(((long long)b * Hh + h) * Mm + m) * NPAD + n] = (half_t)v;
          } else if constexpr (EPI == EPI_SCORES) {
            cb[(long long)r * ldc + c] = (half_t)(v * scale);
          } else if constexpr (EPI == EPI_ATTNO) {
            int b = z / Hh, h = z % Hh;
            ((half_t*)CoutV)[(long long)b * Mm * Hh * Nn + (long long)c * Hh * Nn +
                             h * Nn + r] = (half_t)v;
          } else if constexpr (EPI == EPI_OW1T) {
            v += bias2[r * ldb2 + c];
            ((half_t*)CoutV)[(long long)z * Ra * Ncols + (long long)c * Ra + r] = (half_t)v;
          } else if constexpr (EPI == EPI_OW2) {
            v += bias2[(r % modrow) * ldb2 + c] + (float)resid[(long long)r * ldc + c];
            cb[(long long)r * ldc + c] = (half_t)v;
          } else if constexpr (EPI == EPI_GELU) {
            cb[(long long)r * ldc + c] = (half_t)gelu_exact(v + bias1[c]);
          } else if constexpr (EPI == EPI_GELU_RES) {
            cb[(long long)r * ldc + c] =
                (half_t)(gelu_exact(v + bias1[c]) + (float)resid[(long long)r * ldc + c]);
          } else if constexpr (EPI == EPI_BIAS1) {
            ((float*)CoutV)[(long long)r * ldc + c] = v + bias1[c];
          }
        }
      }
    }
  }
}

// ---------------------------------------------------------------------------
// weight transpose + fp32 -> f16:  in [K x Nc] row-major -> out [Nc x K]
// ---------------------------------------------------------------------------
__global__ __launch_bounds__(256)
void transpose_cvt(const float* __restrict__ in, half_t* __restrict__ out,
                   int K, int Nc, long long tot) {
  for (long long idx = (long long)blockIdx.x * blockDim.x + threadIdx.x; idx < tot;
       idx += (long long)gridDim.x * blockDim.x) {
    long long n = idx / K;
    long long k = idx % K;
    out[idx] = (half_t)in[k * Nc + n];
  }
}

// ---------------------------------------------------------------------------
// patch extraction: [B,224,224,3] fp32 -> [B*196, 768] f16
// ---------------------------------------------------------------------------
__global__ __launch_bounds__(256)
void patch_gather(const float* __restrict__ img, half_t* __restrict__ out,
                  long long total) {
  for (long long idx = (long long)blockIdx.x * blockDim.x + threadIdx.x; idx < total;
       idx += (long long)gridDim.x * blockDim.x) {
    int j  = (int)(idx % 768);
    long long t = idx / 768;
    int n  = (int)(t % 196);
    int b  = (int)(t / 196);
    int c  = j % 3;
    int px = (j / 3) % 16;
    int py = j / 48;
    int pc = n % 14;
    int pr = n / 14;
    out[idx] = (half_t)img[(((long long)b * 224 + pr * 16 + py) * 224 +
                            (pc * 16 + px)) * 3 + c];
  }
}

// ---------------------------------------------------------------------------
// whole-tensor LayerNorm (deterministic tree reduction, fp32 stats, f16 data)
// ---------------------------------------------------------------------------
__global__ __launch_bounds__(256)
void reduce_partials(const half_t* __restrict__ x, long long n,
                     float* __restrict__ part) {
  __shared__ float sh1[256], sh2[256];
  int tid = threadIdx.x;
  float s = 0.f, ss = 0.f;
  for (long long i = (long long)blockIdx.x * 256 + tid; i < n;
       i += (long long)gridDim.x * 256) {
    float v = (float)x[i]; s += v; ss += v * v;
  }
  sh1[tid] = s; sh2[tid] = ss; __syncthreads();
  for (int st = 128; st > 0; st >>= 1) {
    if (tid < st) { sh1[tid] += sh1[tid + st]; sh2[tid] += sh2[tid + st]; }
    __syncthreads();
  }
  if (tid == 0) { part[blockIdx.x] = sh1[0]; part[gridDim.x + blockIdx.x] = sh2[0]; }
}

__global__ __launch_bounds__(256)
void finalize_stats(const float* __restrict__ part, int nb, float invn,
                    float* __restrict__ stats) {
  __shared__ float sh1[256], sh2[256];
  int tid = threadIdx.x;
  float s = 0.f, ss = 0.f;
  for (int i = tid; i < nb; i += 256) { s += part[i]; ss += part[nb + i]; }
  sh1[tid] = s; sh2[tid] = ss; __syncthreads();
  for (int st = 128; st > 0; st >>= 1) {
    if (tid < st) { sh1[tid] += sh1[tid + st]; sh2[tid] += sh2[tid + st]; }
    __syncthreads();
  }
  if (tid == 0) {
    float m   = sh1[0] * invn;
    float var = sh2[0] * invn - m * m;
    stats[0] = m;
    stats[1] = rsqrtf(var + 1e-6f);
  }
}

__global__ __launch_bounds__(256)
void ln_apply(const half_t* __restrict__ x, const float* __restrict__ stats,
              half_t* __restrict__ y, long long n) {
  float m = stats[0], r = stats[1];
  for (long long i = (long long)blockIdx.x * blockDim.x + threadIdx.x; i < n;
       i += (long long)gridDim.x * blockDim.x)
    y[i] = (half_t)(((float)x[i] - m) * r);
}

// ---------------------------------------------------------------------------
// row softmax over 196 (pitch NPAD), one wave32 per row
// ---------------------------------------------------------------------------
__global__ __launch_bounds__(256)
void softmax196(half_t* __restrict__ s, int rows) {
  int row  = blockIdx.x * 8 + (threadIdx.x >> 5);
  int lane = threadIdx.x & 31;
  if (row >= rows) return;
  half_t* p = s + (long long)row * NPAD;
  float vals[7];
  float mx = -3.4e38f;
  #pragma unroll
  for (int j = 0; j < 7; ++j) {
    int idx = lane + 32 * j;
    float v = (idx < 196) ? (float)p[idx] : -3.4e38f;
    vals[j] = v;
    mx = fmaxf(mx, v);
  }
  #pragma unroll
  for (int off = 16; off > 0; off >>= 1) mx = fmaxf(mx, __shfl_xor(mx, off, 32));
  float sum = 0.f;
  #pragma unroll
  for (int j = 0; j < 7; ++j) {
    int idx = lane + 32 * j;
    float e = (idx < 196) ? expf(vals[j] - mx) : 0.f;
    vals[j] = e;
    sum += e;
  }
  #pragma unroll
  for (int off = 16; off > 0; off >>= 1) sum += __shfl_xor(sum, off, 32);
  float inv = 1.0f / sum;
  #pragma unroll
  for (int j = 0; j < 7; ++j) {
    int idx = lane + 32 * j;
    if (idx < 196) p[idx] = (half_t)(vals[j] * inv);
  }
}

// ---------------------------------------------------------------------------
// host orchestration
// ---------------------------------------------------------------------------
#define GEMM(EPI, batch, Ra_, K_, Nc_, A_, lda_, sA_, B_, ldb_, sB_, C_, ldc_, sC_, \
             b1_, b2_, mr_, ldb2_, res_, sc_, Hh_, Mm_, Nn_)                         \
  do {                                                                               \
    dim3 g__(((Nc_) + 127) / 128, ((Ra_) + 127) / 128, (batch));                     \
    gemm_wmma<EPI><<<g__, 256, 0, stream>>>(                                         \
        A_, B_, C_, b1_, b2_, res_, Ra_, K_, Nc_, lda_, (long long)(sA_), ldb_,      \
        (long long)(sB_), ldc_, (long long)(sC_), mr_, ldb2_, sc_, Hh_, Mm_, Nn_);   \
  } while (0)

extern "C" void kernel_launch(void* const* d_in, const int* in_sizes, int n_in,
                              void* d_out, int out_size, void* d_ws, size_t ws_size,
                              hipStream_t stream) {
  const int B = 32, N = 196, D = 768, H = 12, M = 192, L = 6, FF = 3072,
            HEAD = 512, NCLS = 1000;
  const int BN = B * N, MH = M * H, HN = H * N, ND = N * D;
  const long long NTOT = (long long)BN * D;
  const float scale = 1.0f / sqrtf((float)D);

  const float* image   = (const float*)d_in[0];
  const float* patch_w = (const float*)d_in[1];
  const float* patch_b = (const float*)d_in[2];
  const float* pos_emb = (const float*)d_in[3];
  const float* qw = (const float*)d_in[4];   const float* qbv = (const float*)d_in[5];
  const float* kw = (const float*)d_in[6];   const float* kbv = (const float*)d_in[7];
  const float* vw = (const float*)d_in[8];   const float* vbv = (const float*)d_in[9];
  const float* ow1 = (const float*)d_in[10]; const float* ob1 = (const float*)d_in[11];
  const float* ow2 = (const float*)d_in[12]; const float* ob2 = (const float*)d_in[13];
  const float* d1w = (const float*)d_in[14]; const float* d1b = (const float*)d_in[15];
  const float* d2w = (const float*)d_in[16]; const float* d2b = (const float*)d_in[17];
  const float* head_w = (const float*)d_in[18]; const float* head_b = (const float*)d_in[19];
  const float* cls_w  = (const float*)d_in[20]; const float* cls_b  = (const float*)d_in[21];
  float* outp = (float*)d_out;
  (void)in_sizes; (void)n_in; (void)out_size; (void)ws_size;

  // ---- workspace layout (f16 weights/activations + fp32 stats), ~520 MB ----
  char* wsb = (char*)d_ws;
  size_t off = 0;
  auto allocH = [&](size_t n) {
    half_t* p = (half_t*)(wsb + off);
    off = (off + n * sizeof(half_t) + 255) & ~(size_t)255;
    return p;
  };
  auto allocF = [&](size_t n) {
    float* p = (float*)(wsb + off);
    off = (off + n * sizeof(float) + 255) & ~(size_t)255;
    return p;
  };
  half_t* patch_wT = allocH((size_t)D * D);
  half_t* qwT  = allocH((size_t)L * MH * D);
  half_t* kwT  = allocH((size_t)L * MH * D);
  half_t* vwT  = allocH((size_t)L * MH * D);
  half_t* ow1T = allocH((size_t)L * N * HN);
  half_t* ow2T = allocH((size_t)L * D * M);
  half_t* d1wT = allocH((size_t)L * FF * D);
  half_t* d2wT = allocH((size_t)L * D * FF);
  half_t* head_wT = allocH((size_t)HEAD * ND);
  half_t* cls_wT  = allocH((size_t)NCLS * HEAD);
  half_t* patchesH = allocH((size_t)BN * D);
  half_t* xH   = allocH((size_t)BN * D);
  half_t* x1H  = allocH((size_t)BN * D);
  half_t* x2H  = allocH((size_t)BN * D);
  half_t* qH   = allocH((size_t)B * H * N * M);
  half_t* kH   = allocH((size_t)B * H * N * M);
  half_t* vtH  = allocH((size_t)B * H * M * NPAD);
  half_t* scH  = allocH((size_t)B * H * N * NPAD);
  half_t* oH   = allocH((size_t)B * M * HN);
  half_t* tTH  = allocH((size_t)BN * M);
  half_t* hbH  = allocH((size_t)BN * FF);
  half_t* featH = allocH((size_t)B * HEAD);
  float* part  = allocF(1024);
  float* stats = allocF(2);

  auto TR = [&](const float* in, half_t* out, int K, int Nc) {
    long long tot = (long long)K * Nc;
    transpose_cvt<<<2048, 256, 0, stream>>>(in, out, K, Nc, tot);
  };
  auto LN = [&](const half_t* in, half_t* out) {
    reduce_partials<<<512, 256, 0, stream>>>(in, NTOT, part);
    finalize_stats<<<1, 256, 0, stream>>>(part, 512, 1.0f / (float)NTOT, stats);
    ln_apply<<<2048, 256, 0, stream>>>(in, stats, out, NTOT);
  };

  // ---- weight transpose + f16 convert (once per call) ----
  TR(patch_w, patch_wT, D, D);
  for (int l = 0; l < L; ++l) {
    TR(qw + (size_t)l * D * MH, qwT + (size_t)l * MH * D, D, MH);
    TR(kw + (size_t)l * D * MH, kwT + (size_t)l * MH * D, D, MH);
    TR(vw + (size_t)l * D * MH, vwT + (size_t)l * MH * D, D, MH);
    TR(ow1 + (size_t)l * HN * N, ow1T + (size_t)l * N * HN, HN, N);
    TR(ow2 + (size_t)l * M * D, ow2T + (size_t)l * D * M, M, D);
    TR(d1w + (size_t)l * D * FF, d1wT + (size_t)l * FF * D, D, FF);
    TR(d2w + (size_t)l * FF * D, d2wT + (size_t)l * D * FF, FF, D);
  }
  TR(head_w, head_wT, ND, HEAD);
  TR(cls_w, cls_wT, HEAD, NCLS);

  // ---- patch extraction + embedding + positional ----
  patch_gather<<<2048, 256, 0, stream>>>(image, patchesH, NTOT);
  GEMM(EPI_PATCH, 1, BN, D, D, patchesH, D, 0, patch_wT, D, 0, xH, D, 0,
       patch_b, pos_emb, N, D, nullptr, 0.f, 0, 0, 0);

  for (int l = 0; l < L; ++l) {
    const half_t* qwTl = qwT + (size_t)l * MH * D;
    const half_t* kwTl = kwT + (size_t)l * MH * D;
    const half_t* vwTl = vwT + (size_t)l * MH * D;
    const half_t* ow1Tl = ow1T + (size_t)l * N * HN;
    const half_t* ow2Tl = ow2T + (size_t)l * D * M;
    const half_t* d1wTl = d1wT + (size_t)l * FF * D;
    const half_t* d2wTl = d2wT + (size_t)l * D * FF;
    const float* qbl = qbv + (size_t)l * MH;
    const float* kbl = kbv + (size_t)l * MH;
    const float* vbl = vbv + (size_t)l * MH;
    const float* ob1l = ob1 + (size_t)l * M * N;
    const float* ob2l = ob2 + (size_t)l * N * D;
    const float* d1bl = d1b + (size_t)l * FF;
    const float* d2bl = d2b + (size_t)l * D;

    LN(xH, x1H);

    // Q,K -> [B,H,N,M]; V -> [B,H,M,NPAD] (transposed, padded)
    GEMM(EPI_QKV, 1, BN, D, MH, x1H, D, 0, qwTl, D, 0, qH, 0, 0,
         qbl, nullptr, 0, 0, nullptr, 0.f, H, M, N);
    GEMM(EPI_QKV, 1, BN, D, MH, x1H, D, 0, kwTl, D, 0, kH, 0, 0,
         kbl, nullptr, 0, 0, nullptr, 0.f, H, M, N);
    GEMM(EPI_QKVT, 1, BN, D, MH, x1H, D, 0, vwTl, D, 0, vtH, 0, 0,
         vbl, nullptr, 0, 0, nullptr, 0.f, H, M, N);

    // scores[z] = Q[z] * K[z]^T * scale  (K already row-major [N,M] = B^T)
    GEMM(EPI_SCORES, B * H, N, M, N, qH, M, (long long)N * M, kH, M,
         (long long)N * M, scH, NPAD, (long long)N * NPAD, nullptr, nullptr,
         0, 0, nullptr, scale, 0, 0, 0);

    softmax196<<<(B * H * N + 7) / 8, 256, 0, stream>>>(scH, B * H * N);

    // o[b, m, h*N+n] = sum_N' A[z][n,N'] V[z][N',m]  (B^T = VT[z] [M,NPAD])
    GEMM(EPI_ATTNO, B * H, N, N, M, scH, NPAD, (long long)N * NPAD, vtH, NPAD,
         (long long)M * NPAD, oH, 0, 0, nullptr, nullptr, 0, 0, nullptr, 0.f,
         H, M, N);

    // tT[b, n2, m] = (o[b] @ ow1 + ob1)^T, batched over b, shared weight
    GEMM(EPI_OW1T, B, M, HN, N, oH, HN, (long long)M * HN, ow1Tl, HN, 0,
         tTH, 0, 0, nullptr, ob1l, M, N, nullptr, 0.f, 0, 0, 0);

    // x2 = tT @ ow2 + ob2 + x
    GEMM(EPI_OW2, 1, BN, M, D, tTH, M, 0, ow2Tl, M, 0, x2H, D, 0,
         nullptr, ob2l, N, D, xH, 0.f, 0, 0, 0);

    LN(x2H, x1H);   // x3

    // hb = gelu(x3 @ d1w + d1b)
    GEMM(EPI_GELU, 1, BN, D, FF, x1H, D, 0, d1wTl, D, 0, hbH, FF, 0,
         d1bl, nullptr, 0, 0, nullptr, 0.f, 0, 0, 0);

    // x = gelu(hb @ d2w + d2b) + x2
    GEMM(EPI_GELU_RES, 1, BN, FF, D, hbH, FF, 0, d2wTl, FF, 0, xH, D, 0,
         d2bl, nullptr, 0, 0, x2H, 0.f, 0, 0, 0);
  }

  // final LN, head (gelu), classifier (fp32 out to d_out)
  LN(xH, x1H);
  GEMM(EPI_GELU, 1, B, ND, HEAD, x1H, ND, 0, head_wT, ND, 0, featH, HEAD, 0,
       head_b, nullptr, 0, 0, nullptr, 0.f, 0, 0, 0);
  GEMM(EPI_BIAS1, 1, B, HEAD, NCLS, featH, HEAD, 0, cls_wT, HEAD, 0, outp,
       NCLS, 0, cls_b, nullptr, 0, 0, nullptr, 0.f, 0, 0, 0);
}
